// Block_2388001816768
// MI455X (gfx1250) — matrix operand order repaired and checked by reading, not compiled, not run on hardware
//
#include <hip/hip_runtime.h>
#include <hip/hip_bf16.h>
#include <math.h>

#define B_  4
#define T_  2048
#define D_  1024
#define H_  16
#define HD_ 64
#define FF_ 4096
#define M_  (B_*T_)   // 8192 rows

typedef __attribute__((ext_vector_type(16))) __bf16 v16bf;
typedef __attribute__((ext_vector_type(8)))  float  v8f;
typedef __attribute__((ext_vector_type(2)))  int    v2i_t;

union AFrag { v16bf v; uint4 u[2]; };

static __device__ __forceinline__ v8f wmma_bf16(v16bf a, v16bf b, v8f c) {
  // v_wmma_f32_16x16x32_bf16, f32 accumulate
  return __builtin_amdgcn_wmma_f32_16x16x32_bf16(false, a, false, b, (short)0, c, false, false);
}

// 16-lane max reduction, pure VALU via DPP16 (quad_perm xor1/xor2, half-mirror, mirror).
static __device__ __forceinline__ float redmax16(float x) {
  int t;
  t = __builtin_amdgcn_update_dpp(0, __builtin_bit_cast(int, x), 0xB1, 0xF, 0xF, true);  // quad_perm [1,0,3,2]
  x = fmaxf(x, __builtin_bit_cast(float, t));
  t = __builtin_amdgcn_update_dpp(0, __builtin_bit_cast(int, x), 0x4E, 0xF, 0xF, true);  // quad_perm [2,3,0,1]
  x = fmaxf(x, __builtin_bit_cast(float, t));
  t = __builtin_amdgcn_update_dpp(0, __builtin_bit_cast(int, x), 0x141, 0xF, 0xF, true); // row_half_mirror
  x = fmaxf(x, __builtin_bit_cast(float, t));
  t = __builtin_amdgcn_update_dpp(0, __builtin_bit_cast(int, x), 0x140, 0xF, 0xF, true); // row_mirror
  x = fmaxf(x, __builtin_bit_cast(float, t));
  return x;
}

#if __has_builtin(__builtin_amdgcn_global_load_async_to_lds_b64)
#define HAVE_ASYNC_LDS 1
#else
#define HAVE_ASYNC_LDS 0
#endif

#if __has_builtin(__builtin_amdgcn_s_wait_asynccnt)
#define WAIT_ASYNC(n) __builtin_amdgcn_s_wait_asynccnt(n)
#else
#define WAIT_ASYNC(n) asm volatile("s_wait_asynccnt %0" :: "i"(n) : "memory")
#endif

// ---------------- weight repack (one-time, tiny vs GEMMs) ----------------

__global__ __launch_bounds__(256) void conv_qkv_w(const float* __restrict__ Wq,
                                                  const float* __restrict__ Wk,
                                                  const float* __restrict__ Wv,
                                                  __bf16* __restrict__ Wt) {
  const int idx = blockIdx.x * 256 + threadIdx.x;      // 3*D*D total
  const int n = idx >> 10;
  const int d = idx & (D_ - 1);
  const float* W = (n < D_) ? Wq : (n < 2 * D_) ? Wk : Wv;
  const int nn = n & (D_ - 1);
  const int hh = nn >> 6, k = nn & 63;
  Wt[(size_t)n * D_ + d] = (__bf16)W[((size_t)hh * D_ + d) * HD_ + k];
}

__global__ __launch_bounds__(256) void conv_transpose(const float* __restrict__ W,
                                                      __bf16* __restrict__ Wt,
                                                      int K, int N) {
  const int idx = blockIdx.x * 256 + threadIdx.x;      // K*N total
  const int n = idx % N;
  const int d = idx / N;
  Wt[(size_t)n * K + d] = (__bf16)W[(size_t)d * N + n];
}

// ---------------- LayerNorm -> bf16 activations ----------------

__global__ __launch_bounds__(256) void ln_rows(const float* __restrict__ x,
                                               const float* __restrict__ gw,
                                               const float* __restrict__ bw,
                                               __bf16* __restrict__ out) {
  const int row = blockIdx.x;
  const float* xr = x + (size_t)row * D_;
  const int t = threadIdx.x;
  float4 v = *(const float4*)(xr + t * 4);
  __shared__ float red[256];
  red[t] = v.x + v.y + v.z + v.w;
  __syncthreads();
  for (int o = 128; o > 0; o >>= 1) { if (t < o) red[t] += red[t + o]; __syncthreads(); }
  const float mean = red[0] * (1.0f / D_);
  __syncthreads();
  const float d0 = v.x - mean, d1 = v.y - mean, d2 = v.z - mean, d3 = v.w - mean;
  red[t] = d0 * d0 + d1 * d1 + d2 * d2 + d3 * d3;
  __syncthreads();
  for (int o = 128; o > 0; o >>= 1) { if (t < o) red[t] += red[t + o]; __syncthreads(); }
  const float rstd = rsqrtf(red[0] * (1.0f / D_) + 1e-5f);
  const float4 gg = *(const float4*)(gw + t * 4);
  const float4 bb = *(const float4*)(bw + t * 4);
  __bf16* orow = out + (size_t)row * D_ + t * 4;
  orow[0] = (__bf16)(d0 * rstd * gg.x + bb.x);
  orow[1] = (__bf16)(d1 * rstd * gg.y + bb.y);
  orow[2] = (__bf16)(d2 * rstd * gg.z + bb.z);
  orow[3] = (__bf16)(d3 * rstd * gg.w + bb.w);
}

// ---------------- WMMA GEMM: block = 32(M) x 512(N), wave = 32x64 ----------------
// A panel (32 x K) double-buffer staged through LDS with CDNA5 async loads.
// EPI 0: QKV split (bias, Q scaled 1/sqrt(HD); Q/K -> [B,H,T,HD] bf16, V -> [B,H,HD,T] bf16)
// EPI 1: bias + f32 residual -> f32 out
// EPI 2: bias + exact-erf GELU -> bf16 out

template <int N, int K, int EPI>
__global__ __launch_bounds__(256) void gemm_bf16_wmma(
    const __bf16* __restrict__ A, const __bf16* __restrict__ Bt,
    const float* __restrict__ c0, const float* __restrict__ c1, const float* __restrict__ c2,
    const float* __restrict__ resid, float* __restrict__ outF,
    __bf16* __restrict__ o0, __bf16* __restrict__ o1, __bf16* __restrict__ o2) {
  constexpr int NB = N / 512;
  const int m0 = (blockIdx.x / NB) * 32;
  const int warp = threadIdx.x >> 5;
  const int n0 = (blockIdx.x % NB) * 512 + warp * 64;
  const int lane = threadIdx.x & 31;
  const int g = lane >> 4, ln = lane & 15;

  __shared__ alignas(16) __bf16 Asm[2][32 * 32];   // double-buffered 32x32 A tile (2KB each)

  const int tid = threadIdx.x;
  const int rr = tid >> 3;             // 0..31 : A row
  const int cc = (tid & 7) * 4;        // 0..28 : A col (4 bf16 = 8B per thread)
  const __bf16* Astage = A + (size_t)(m0 + rr) * K + cc;

  auto stage = [&](int buf, int kk) {
    const __bf16* src = Astage + kk;
    __bf16* dst = &Asm[buf][rr * 32 + cc];
#if HAVE_ASYNC_LDS
    typedef __attribute__((address_space(1))) v2i_t* gptr_t;
    typedef __attribute__((address_space(3))) v2i_t* lptr_t;
    __builtin_amdgcn_global_load_async_to_lds_b64(
        (gptr_t)(void*)const_cast<__bf16*>(src),
        (lptr_t)(void*)dst, 0, 0);
#else
    *(uint2*)dst = *(const uint2*)src;
#endif
  };

  stage(0, 0);

  v8f acc[2][4] = {};
  for (int kk = 0; kk < K; kk += 32) {
    const int cur = (kk >> 5) & 1;
    const bool more = (kk + 32) < K;
    if (more) stage(cur ^ 1, kk + 32);
#if HAVE_ASYNC_LDS
    if (more) WAIT_ASYNC(1);            // current buffer landed, next still in flight
    else      WAIT_ASYNC(0);
#endif
    __syncthreads();

    AFrag a0, a1;
    const __bf16* Ar0 = &Asm[cur][ln * 32];
    const __bf16* Ar1 = &Asm[cur][(16 + ln) * 32];
    a0.u[0] = *(const uint4*)(Ar0 + g * 16);
    a0.u[1] = *(const uint4*)(Ar0 + 8 + g * 16);
    a1.u[0] = *(const uint4*)(Ar1 + g * 16);
    a1.u[1] = *(const uint4*)(Ar1 + 8 + g * 16);

    const __bf16* bp = Bt + (size_t)(n0 + ln) * K + kk + g * 16;
#pragma unroll
    for (int nb = 0; nb < 4; ++nb) {
      const __bf16* bpn = bp + (size_t)(nb * 16) * K;
      __builtin_prefetch(bpn + 64, 0, 0);                        // global_prefetch_b8, 2 steps ahead
      v16bf b = *(const v16bf*)bpn;                              // contiguous 32B per lane
      acc[0][nb] = wmma_bf16(a0.v, b, acc[0][nb]);
      acc[1][nb] = wmma_bf16(a1.v, b, acc[1][nb]);
    }
    __syncthreads();                                             // reads done before buf reuse
  }

#pragma unroll
  for (int mt = 0; mt < 2; ++mt) {
#pragma unroll
    for (int nb = 0; nb < 4; ++nb) {
#pragma unroll
      for (int r = 0; r < 8; ++r) {
        const int m = m0 + mt * 16 + r + 8 * g;   // C layout: elem r -> M=r+8g, N=lane%16
        const int n = n0 + nb * 16 + ln;
        const float v = acc[mt][nb][r];
        if constexpr (EPI == 0) {
          const int bb = m >> 11, t = m & (T_ - 1);
          if (n < D_) {
            const float q = (v + c0[n]) * 0.125f;               // fold 1/sqrt(64)
            const int hh = n >> 6, k = n & 63;
            o0[(((size_t)(bb * H_ + hh)) * T_ + t) * HD_ + k] = (__bf16)q;
          } else if (n < 2 * D_) {
            const int n2 = n - D_;
            const int hh = n2 >> 6, k = n2 & 63;
            o1[(((size_t)(bb * H_ + hh)) * T_ + t) * HD_ + k] = (__bf16)(v + c1[n2]);
          } else {
            const int n2 = n - 2 * D_;
            const int hh = n2 >> 6, k = n2 & 63;
            o2[((size_t)(bb * H_ + hh) * HD_ + k) * T_ + t] = (__bf16)(v + c2[n2]);  // V^T
          }
        } else if constexpr (EPI == 1) {
          outF[(size_t)m * N + n] = v + c0[n] + resid[(size_t)m * N + n];
        } else {
          const float u = v + c0[n];
          const float ge = 0.5f * u * (1.0f + erff(u * 0.70710678118654752f));
          o0[(size_t)m * N + n] = (__bf16)ge;
        }
      }
    }
  }
}

// ---------------- Flash attention: 16-query tile per wave, causal, online softmax ---------
// Row max via DPP16 butterflies; row sum via one extra WMMA against an all-ones B matrix.

__global__ __launch_bounds__(128) void attn_flash(const __bf16* __restrict__ Q,
                                                  const __bf16* __restrict__ Kk,
                                                  const __bf16* __restrict__ Vt,
                                                  __bf16* __restrict__ ctx) {
  const int warp = threadIdx.x >> 5;
  const int wid = blockIdx.x * 4 + warp;
  const int qt = wid & (T_ / 16 - 1);
  const int h = (wid >> 7) & (H_ - 1);
  const int b = wid >> 11;
  const int q0 = qt * 16;
  const int lane = threadIdx.x & 31;
  const int g = lane >> 4, ln = lane & 15;

  const __bf16* Qb = Q + ((size_t)(b * H_ + h) * T_) * HD_;
  const __bf16* Kb = Kk + ((size_t)(b * H_ + h) * T_) * HD_;
  const __bf16* Vb = Vt + ((size_t)(b * H_ + h) * HD_) * T_;   // [HD, T]

  __shared__ alignas(32) unsigned short Pl[4][16 * 32];        // per-wave P staging
  unsigned short* myP = Pl[warp];

  AFrag aq0, aq1;                       // Q fragments, loop-invariant
  const __bf16* Qrow = Qb + (size_t)(q0 + ln) * HD_;
  aq0.u[0] = *(const uint4*)(Qrow + g * 16);
  aq0.u[1] = *(const uint4*)(Qrow + 8 + g * 16);
  aq1.u[0] = *(const uint4*)(Qrow + 32 + g * 16);
  aq1.u[1] = *(const uint4*)(Qrow + 40 + g * 16);

  v16bf ones;
#pragma unroll
  for (int i = 0; i < 16; ++i) ones[i] = (__bf16)1.0f;

  float mi[8], li[8];
#pragma unroll
  for (int r = 0; r < 8; ++r) { mi[r] = -INFINITY; li[r] = 0.0f; }
  v8f oacc[4] = {};

  const int kv_end = q0 + 16;           // causal: keys <= q0+15 needed
  for (int kv = 0; kv < kv_end; kv += 32) {
    v8f s[2];
#pragma unroll
    for (int st = 0; st < 2; ++st) {    // two 16-key score tiles: S = Q * K^T
      const __bf16* Krow = Kb + (size_t)(kv + st * 16 + ln) * HD_;
      v16bf bk0 = *(const v16bf*)(Krow + g * 16);
      v16bf bk1 = *(const v16bf*)(Krow + 32 + g * 16);
      v8f z = {};
      z = wmma_bf16(aq0.v, bk0, z);
      s[st] = wmma_bf16(aq1.v, bk1, z);
    }

    float scale[8];
#pragma unroll
    for (int r = 0; r < 8; ++r) {       // per query row (rows live across the 16-lane group)
      const int q = q0 + r + 8 * g;
#pragma unroll
      for (int st = 0; st < 2; ++st) {
        const int key = kv + st * 16 + ln;
        if (key > q) s[st][r] = -INFINITY;
      }
      const float mrow = redmax16(fmaxf(s[0][r], s[1][r]));
      const float mnew = fmaxf(mi[r], mrow);
      scale[r] = __expf(mi[r] - mnew);
      mi[r] = mnew;
#pragma unroll
      for (int st = 0; st < 2; ++st) {
        const float pe = __expf(s[st][r] - mnew);
        const __bf16 pb = (__bf16)pe;
        myP[(r + 8 * g) * 32 + st * 16 + ln] = __builtin_bit_cast(unsigned short, pb);
      }
#pragma unroll
      for (int nb = 0; nb < 4; ++nb) oacc[nb][r] *= scale[r];
    }

    __asm__ volatile("s_wait_dscnt 0" ::: "memory");  // same-wave LDS RAW
    AFrag ap;
    const __bf16* Prow = (const __bf16*)myP + ln * 32;
    ap.u[0] = *(const uint4*)(Prow + g * 16);
    ap.u[1] = *(const uint4*)(Prow + 8 + g * 16);

    v8f lz = {};
    const v8f lacc = wmma_bf16(ap.v, ones, lz);       // row sums of P, C layout (replicated over N)
#pragma unroll
    for (int r = 0; r < 8; ++r) li[r] = li[r] * scale[r] + lacc[r];

#pragma unroll
    for (int nb = 0; nb < 4; ++nb) {    // ctx += P * V  (V transposed -> contiguous B frags)
      v16bf bv = *(const v16bf*)(Vb + (size_t)(nb * 16 + ln) * T_ + kv + g * 16);
      oacc[nb] = wmma_bf16(ap.v, bv, oacc[nb]);
    }
  }

  float inv[8];
#pragma unroll
  for (int r = 0; r < 8; ++r) inv[r] = 1.0f / li[r];
#pragma unroll
  for (int nb = 0; nb < 4; ++nb)
#pragma unroll
    for (int r = 0; r < 8; ++r)
      ctx[((size_t)(b * T_ + q0 + r + 8 * g)) * D_ + h * HD_ + nb * 16 + ln] =
          (__bf16)(oacc[nb][r] * inv[r]);
}

// ---------------- host-side orchestration ----------------

extern "C" void kernel_launch(void* const* d_in, const int* in_sizes, int n_in,
                              void* d_out, int out_size, void* d_ws, size_t ws_size,
                              hipStream_t stream) {
  const float* x    = (const float*)d_in[0];
  const float* ln1g = (const float*)d_in[2];
  const float* ln1b = (const float*)d_in[3];
  const float* Wq   = (const float*)d_in[4];
  const float* bq   = (const float*)d_in[5];
  const float* Wk   = (const float*)d_in[6];
  const float* bk   = (const float*)d_in[7];
  const float* Wv   = (const float*)d_in[8];
  const float* bv   = (const float*)d_in[9];
  const float* Wo   = (const float*)d_in[10];
  const float* bo   = (const float*)d_in[11];
  const float* ln2g = (const float*)d_in[12];
  const float* ln2b = (const float*)d_in[13];
  const float* W1   = (const float*)d_in[14];
  const float* b1   = (const float*)d_in[15];
  const float* W2   = (const float*)d_in[16];
  const float* b2   = (const float*)d_in[17];
  float* out = (float*)d_out;

  char* ws = (char*)d_ws;
  size_t off = 0;
  auto take = [&](size_t bytes) {
    char* p = ws + off;
    off += (bytes + 255) & ~(size_t)255;
    return (void*)p;
  };
  __bf16* Wqkv_t = (__bf16*)take((size_t)3 * D_ * D_ * 2);
  __bf16* Wo_t   = (__bf16*)take((size_t)D_ * D_ * 2);
  __bf16* W1_t   = (__bf16*)take((size_t)D_ * FF_ * 2);
  __bf16* W2_t   = (__bf16*)take((size_t)D_ * FF_ * 2);
  __bf16* h1     = (__bf16*)take((size_t)M_ * D_ * 2);
  __bf16* Qb     = (__bf16*)take((size_t)M_ * D_ * 2);
  __bf16* Kb     = (__bf16*)take((size_t)M_ * D_ * 2);
  __bf16* Vt     = (__bf16*)take((size_t)M_ * D_ * 2);
  __bf16* ctx    = (__bf16*)take((size_t)M_ * D_ * 2);
  float*  x1     = (float*)take((size_t)M_ * D_ * 4);
  __bf16* h2     = (__bf16*)take((size_t)M_ * D_ * 2);
  __bf16* ff     = (__bf16*)take((size_t)M_ * FF_ * 2);

  // weight repack -> bf16, transposed (Bt layout, contiguous WMMA B frags)
  conv_qkv_w<<<3 * D_ * D_ / 256, 256, 0, stream>>>(Wq, Wk, Wv, Wqkv_t);
  conv_transpose<<<D_ * D_ / 256, 256, 0, stream>>>(Wo, Wo_t, D_, D_);
  conv_transpose<<<D_ * FF_ / 256, 256, 0, stream>>>(W1, W1_t, D_, FF_);
  conv_transpose<<<D_ * FF_ / 256, 256, 0, stream>>>(W2, W2_t, FF_, D_);

  // LN1 -> h1 (bf16)
  ln_rows<<<M_, 256, 0, stream>>>(x, ln1g, ln1b, h1);

  // QKV projection: [8192,1024] x [1024,3072]
  gemm_bf16_wmma<3 * D_, D_, 0>
      <<<(M_ / 32) * (3 * D_ / 512), 256, 0, stream>>>(
          h1, Wqkv_t, bq, bk, bv, nullptr, nullptr, Qb, Kb, Vt);

  // flash attention -> ctx (bf16, heads concatenated)
  attn_flash<<<(B_ * H_ * (T_ / 16)) / 4, 128, 0, stream>>>(Qb, Kb, Vt, ctx);

  // output projection + residual -> x1 (f32)
  gemm_bf16_wmma<D_, D_, 1>
      <<<(M_ / 32) * (D_ / 512), 256, 0, stream>>>(
          ctx, Wo_t, bo, nullptr, nullptr, x, x1, nullptr, nullptr, nullptr);

  // LN2 -> h2 (bf16)
  ln_rows<<<M_, 256, 0, stream>>>(x1, ln2g, ln2b, h2);

  // FFN1 + exact GELU -> ff (bf16)
  gemm_bf16_wmma<FF_, D_, 2>
      <<<(M_ / 32) * (FF_ / 512), 256, 0, stream>>>(
          h2, W1_t, b1, nullptr, nullptr, nullptr, nullptr, ff, nullptr, nullptr);

  // FFN2 + residual -> d_out (f32)
  gemm_bf16_wmma<D_, FF_, 1>
      <<<(M_ / 32) * (D_ / 512), 256, 0, stream>>>(
          ff, W2_t, b2, nullptr, nullptr, x1, out, nullptr, nullptr, nullptr);

  (void)in_sizes; (void)n_in; (void)out_size; (void)ws_size;
}
